// DiffTransformerEncoderLayer_15848429322352
// MI455X (gfx1250) — matrix operand order repaired and checked by reading, not compiled, not run on hardware
//
#include <hip/hip_runtime.h>

#define D_MODEL 512
#define N_HEADS 8
#define DK      64
#define D_FF    2048
#define B_SZ    2
#define L_SEQ   1024
#define DIFF_H  32
#define M_ROWS  (B_SZ * L_SEQ)   // 2048

typedef __attribute__((ext_vector_type(16))) _Float16 v16h;
typedef __attribute__((ext_vector_type(8)))  _Float16 v8h;
typedef __attribute__((ext_vector_type(4)))  _Float16 v4h;
typedef __attribute__((ext_vector_type(8)))  float    v8f;

static __device__ __forceinline__ v16h cat8(v8h lo, v8h hi) {
    return __builtin_shufflevector(lo, hi, 0, 1, 2, 3, 4, 5, 6, 7,
                                           8, 9, 10, 11, 12, 13, 14, 15);
}

// ---------------------------------------------------------------------------
// Tiled transpose + f32->f16 convert:  out[N][K] (f16) = in[K][N] (f32)
// ---------------------------------------------------------------------------
__global__ void __launch_bounds__(256)
transp_f16(const float* __restrict__ in, _Float16* __restrict__ out, int K, int N)
{
    __shared__ float tile[32][33];
    const int tx = threadIdx.x & 31;
    const int ty = threadIdx.x >> 5;          // 0..7
    const int n0 = blockIdx.x * 32;
    const int k0 = blockIdx.y * 32;

    #pragma unroll
    for (int s = 0; s < 4; s++)
        tile[ty + 8 * s][tx] = in[(size_t)(k0 + ty + 8 * s) * N + n0 + tx];
    __syncthreads();
    #pragma unroll
    for (int s = 0; s < 4; s++)
        out[(size_t)(n0 + ty + 8 * s) * K + k0 + tx] = (_Float16)tile[tx][ty + 8 * s];
}

// ---------------------------------------------------------------------------
// Tiled WMMA GEMM: C = act(A[M,K] @ W + bias), W pre-transposed as Wt[N][K] f16.
// block = 128 threads (4 waves), 64x64 tile, K-step 32, double-buffered LDS:
// next k-step's global loads are issued before this step's WMMAs, stored to
// the alternate buffer after, one barrier per iteration.
// outmode: 0 -> f32 Cf ; 1 -> f16 Ch row-major ; 2 -> f16 Ch transposed.
// ---------------------------------------------------------------------------
__global__ void __launch_bounds__(128)
gemm_wmma(const float* __restrict__ A, const _Float16* __restrict__ Wt,
          const float* __restrict__ bias, float* __restrict__ Cf,
          _Float16* __restrict__ Ch,
          int M, int N, int K, int relu, int outmode)
{
    __shared__ __align__(16) _Float16 As[2][64][40];    // row stride 80 B
    __shared__ __align__(16) _Float16 BsT[2][64][40];   // [col][k]

    const int tid  = threadIdx.x;
    const int wave = tid >> 5;
    const int lane = tid & 31;
    const int g16  = lane >> 4;
    const int l16  = lane & 15;
    const int tile_m = blockIdx.y * 64;
    const int tile_n = blockIdx.x * 64;

    // per-thread staging coordinates
    int ar[4], acq[4], bc[2], boct[2];
    #pragma unroll
    for (int it = 0; it < 4; it++) { int i = tid + it * 128; ar[it] = i >> 3; acq[it] = i & 7; }
    #pragma unroll
    for (int it = 0; it < 2; it++) { int i = tid + it * 128; bc[it] = i >> 2; boct[it] = i & 3; }

    v8f acc[4];
    #pragma unroll
    for (int n = 0; n < 4; n++)
        #pragma unroll
        for (int e = 0; e < 8; e++) acc[n][e] = 0.0f;

    const int nk = K >> 5;

    // ---- prologue: stage k-step 0 into buffer 0 (loads batched first) ----
    {
        float4 atmp[4]; v8h btmp[2];
        #pragma unroll
        for (int it = 0; it < 4; it++)
            atmp[it] = *(const float4*)&A[(size_t)(tile_m + ar[it]) * K + acq[it] * 4];
        #pragma unroll
        for (int it = 0; it < 2; it++)
            btmp[it] = *(const v8h*)&Wt[(size_t)(tile_n + bc[it]) * K + boct[it] * 8];
        #pragma unroll
        for (int it = 0; it < 4; it++) {
            v4h hv;
            hv[0] = (_Float16)atmp[it].x; hv[1] = (_Float16)atmp[it].y;
            hv[2] = (_Float16)atmp[it].z; hv[3] = (_Float16)atmp[it].w;
            *(v4h*)&As[0][ar[it]][acq[it] * 4] = hv;
        }
        #pragma unroll
        for (int it = 0; it < 2; it++)
            *(v8h*)&BsT[0][bc[it]][boct[it] * 8] = btmp[it];
    }
    __syncthreads();

    for (int k = 0; k < nk; k++) {
        const int buf = k & 1;
        const bool havenext = (k + 1 < nk);
        float4 atmp[4]; v8h btmp[2];

        // issue next k-step's global loads early (latency hidden by WMMAs)
        if (havenext) {
            const int kk = (k + 1) << 5;
            #pragma unroll
            for (int it = 0; it < 4; it++)
                atmp[it] = *(const float4*)&A[(size_t)(tile_m + ar[it]) * K + kk + acq[it] * 4];
            #pragma unroll
            for (int it = 0; it < 2; it++)
                btmp[it] = *(const v8h*)&Wt[(size_t)(tile_n + bc[it]) * K + kk + boct[it] * 8];
        }
        if (k + 2 < nk)  // prefetch two steps ahead (global_prefetch_b8)
            __builtin_prefetch(&A[(size_t)(tile_m + (tid >> 1)) * K + ((k + 2) << 5)], 0, 1);

        // compute from current buffer
        const int arow = wave * 16 + l16;
        v16h a = cat8(*(const v8h*)&As[buf][arow][g16 * 8],
                      *(const v8h*)&As[buf][arow][16 + g16 * 8]);
        #pragma unroll
        for (int n = 0; n < 4; n++) {
            const int bcol = n * 16 + l16;
            v16h b = cat8(*(const v8h*)&BsT[buf][bcol][g16 * 8],
                          *(const v8h*)&BsT[buf][bcol][16 + g16 * 8]);
            acc[n] = __builtin_amdgcn_wmma_f32_16x16x32_f16(
                false, a, false, b, (short)0, acc[n], false, false);
        }

        // store next step into alternate buffer
        if (havenext) {
            #pragma unroll
            for (int it = 0; it < 4; it++) {
                v4h hv;
                hv[0] = (_Float16)atmp[it].x; hv[1] = (_Float16)atmp[it].y;
                hv[2] = (_Float16)atmp[it].z; hv[3] = (_Float16)atmp[it].w;
                *(v4h*)&As[buf ^ 1][ar[it]][acq[it] * 4] = hv;
            }
            #pragma unroll
            for (int it = 0; it < 2; it++)
                *(v8h*)&BsT[buf ^ 1][bc[it]][boct[it] * 8] = btmp[it];
        }
        __syncthreads();
    }

    // C layout: VGPR r, lane -> row = r + 8*half, col = lane%16
    #pragma unroll
    for (int n = 0; n < 4; n++) {
        #pragma unroll
        for (int r = 0; r < 8; r++) {
            int row = tile_m + wave * 16 + r + g16 * 8;
            int col = tile_n + n * 16 + l16;
            float v = acc[n][r] + bias[col];
            if (relu) v = fmaxf(v, 0.0f);
            if (outmode == 0)      Cf[(size_t)row * N + col] = v;
            else if (outmode == 1) Ch[(size_t)row * N + col] = (_Float16)v;
            else                   Ch[(size_t)col * M + row] = (_Float16)v;
        }
    }
}

// ---------------------------------------------------------------------------
// Fused attention: one wave per (b, h, 16-row q tile).
// Q,K in f16 (B,L,D); V in f16 transposed Vt[d][b*L + k].
// ---------------------------------------------------------------------------
__global__ void __launch_bounds__(32)
attn_wmma(const _Float16* __restrict__ Q, const _Float16* __restrict__ Kp,
          const _Float16* __restrict__ Vt, const float* __restrict__ mz,
          const unsigned char* __restrict__ pad,
          const float* __restrict__ dw1, const float* __restrict__ db1,
          const float* __restrict__ dw2, const float* __restrict__ db2p,
          float* __restrict__ ctx)
{
    __shared__ __align__(16) _Float16 Pst[16][32];   // P restage, stride 64 B

    const int lane = threadIdx.x;
    const int g16  = lane >> 4;
    const int l16  = lane & 15;

    int bid = blockIdx.x;
    const int qt = bid & 63;  bid >>= 6;     // L/16 = 64 q tiles
    const int h  = bid & 7;   bid >>= 3;
    const int b  = bid;
    const int qbase = qt * 16;
    const float db2 = db2p[0];

    const _Float16* qp = Q + ((size_t)(b * L_SEQ + qbase + l16)) * D_MODEL + h * DK;
    v16h qa0 = cat8(*(const v8h*)&qp[g16 * 8],      *(const v8h*)&qp[16 + g16 * 8]);
    v16h qa1 = cat8(*(const v8h*)&qp[32 + g16 * 8], *(const v8h*)&qp[48 + g16 * 8]);

    float mzq[8];
    #pragma unroll
    for (int r = 0; r < 8; r++)
        mzq[r] = mz[b * L_SEQ + qbase + r + g16 * 8];

    v8f vc[4];
    #pragma unroll
    for (int n = 0; n < 4; n++)
        #pragma unroll
        for (int e = 0; e < 8; e++) vc[n][e] = 0.0f;
    float mrow[8], lrow[8];
    #pragma unroll
    for (int r = 0; r < 8; r++) { mrow[r] = -3.0e38f; lrow[r] = 0.0f; }

    for (int kb = 0; kb < L_SEQ; kb += 32) {
        // ---- scores for 32 key columns (two 16x16 C tiles) ----
        v8f S[2];
        #pragma unroll
        for (int c = 0; c < 2; c++) {
            const _Float16* kp = Kp + ((size_t)(b * L_SEQ + kb + c * 16 + l16)) * D_MODEL + h * DK;
            v16h kb0 = cat8(*(const v8h*)&kp[g16 * 8],      *(const v8h*)&kp[16 + g16 * 8]);
            v16h kb1 = cat8(*(const v8h*)&kp[32 + g16 * 8], *(const v8h*)&kp[48 + g16 * 8]);
            v8f s;
            #pragma unroll
            for (int e = 0; e < 8; e++) s[e] = 0.0f;
            s = __builtin_amdgcn_wmma_f32_16x16x32_f16(false, qa0, false, kb0, (short)0, s, false, false);
            s = __builtin_amdgcn_wmma_f32_16x16x32_f16(false, qa1, false, kb1, (short)0, s, false, false);
            S[c] = s;
        }

        // ---- scale + diff-MLP + pad mask ----
        float mzk[2]; int kcol[2], padk[2];
        #pragma unroll
        for (int c = 0; c < 2; c++) {
            kcol[c] = kb + c * 16 + l16;
            mzk[c]  = mz[b * L_SEQ + kcol[c]];
            padk[c] = pad[b * L_SEQ + kcol[c]];
        }
        #pragma unroll
        for (int c = 0; c < 2; c++) {
            #pragma unroll
            for (int r = 0; r < 8; r++) {
                int qrow = qbase + r + g16 * 8;
                float s = S[c][r] * 0.125f;       // 1/sqrt(64)
                float d = (qrow == 0 || kcol[c] == 0) ? 0.0f : fabsf(mzq[r] - mzk[c]);
                float accm = db2;
                for (int j = 0; j < DIFF_H; j++) {
                    float hh = fmaf(d, dw1[j], db1[j]);
                    hh = fmaxf(hh, 0.0f);
                    accm = fmaf(hh, dw2[j], accm);
                }
                s += accm;
                if (padk[c]) s = -3.0e38f;
                S[c][r] = s;
            }
        }

        // ---- online softmax update ----
        #pragma unroll
        for (int r = 0; r < 8; r++) {
            float tmax = fmaxf(S[0][r], S[1][r]);
            for (int m = 1; m < 16; m <<= 1) tmax = fmaxf(tmax, __shfl_xor(tmax, m, 32));
            float mnew  = fmaxf(mrow[r], tmax);
            float scale = __expf(mrow[r] - mnew);
            float p0 = __expf(S[0][r] - mnew);
            float p1 = __expf(S[1][r] - mnew);
            float tsum = p0 + p1;
            for (int m = 1; m < 16; m <<= 1) tsum += __shfl_xor(tsum, m, 32);
            lrow[r] = lrow[r] * scale + tsum;
            mrow[r] = mnew;
            #pragma unroll
            for (int n = 0; n < 4; n++) vc[n][r] *= scale;
            S[0][r] = p0;
            S[1][r] = p1;
        }

        // ---- restage P (C-layout -> A-layout) through LDS ----
        __syncthreads();
        #pragma unroll
        for (int c = 0; c < 2; c++)
            #pragma unroll
            for (int r = 0; r < 8; r++)
                Pst[r + g16 * 8][c * 16 + l16] = (_Float16)S[c][r];
        __syncthreads();

        v16h pa = cat8(*(const v8h*)&Pst[l16][g16 * 8],
                       *(const v8h*)&Pst[l16][16 + g16 * 8]);

        // ---- ctx += P @ V ----
        #pragma unroll
        for (int n = 0; n < 4; n++) {
            const _Float16* vp = Vt + (size_t)(h * DK + n * 16 + l16) * M_ROWS
                                 + b * L_SEQ + kb + g16 * 8;
            v16h vb = cat8(*(const v8h*)&vp[0], *(const v8h*)&vp[16]);
            vc[n] = __builtin_amdgcn_wmma_f32_16x16x32_f16(
                false, pa, false, vb, (short)0, vc[n], false, false);
        }
        __syncthreads();
    }

    // ---- normalize and write ctx (B,L,D) f32 ----
    #pragma unroll
    for (int n = 0; n < 4; n++) {
        #pragma unroll
        for (int r = 0; r < 8; r++) {
            size_t row = (size_t)(b * L_SEQ + qbase + r + g16 * 8);
            ctx[row * D_MODEL + h * DK + n * 16 + l16] = vc[n][r] / lrow[r];
        }
    }
}

// ---------------------------------------------------------------------------
// out = LayerNorm(a + b) * g + beta
// ---------------------------------------------------------------------------
__global__ void __launch_bounds__(256)
add_ln(const float* __restrict__ a, const float* __restrict__ bR,
       const float* __restrict__ g, const float* __restrict__ beta,
       float* __restrict__ out)
{
    __shared__ float red[256];
    const int row = blockIdx.x;
    const int tid = threadIdx.x;
    const float* ap = a  + (size_t)row * D_MODEL;
    const float* bp = bR + (size_t)row * D_MODEL;

    float v0 = ap[tid] + bp[tid];
    float v1 = ap[tid + 256] + bp[tid + 256];

    red[tid] = v0 + v1;
    __syncthreads();
    for (int o = 128; o > 0; o >>= 1) { if (tid < o) red[tid] += red[tid + o]; __syncthreads(); }
    float mean = red[0] * (1.0f / D_MODEL);
    __syncthreads();

    float d0 = v0 - mean, d1 = v1 - mean;
    red[tid] = d0 * d0 + d1 * d1;
    __syncthreads();
    for (int o = 128; o > 0; o >>= 1) { if (tid < o) red[tid] += red[tid + o]; __syncthreads(); }
    float inv = rsqrtf(red[0] * (1.0f / D_MODEL) + 1e-5f);

    out[(size_t)row * D_MODEL + tid]       = d0 * inv * g[tid]       + beta[tid];
    out[(size_t)row * D_MODEL + tid + 256] = d1 * inv * g[tid + 256] + beta[tid + 256];
}

// ---------------------------------------------------------------------------
extern "C" void kernel_launch(void* const* d_in, const int* in_sizes, int n_in,
                              void* d_out, int out_size, void* d_ws, size_t ws_size,
                              hipStream_t stream)
{
    const float* x   = (const float*)d_in[0];
    const float* mz  = (const float*)d_in[1];
    const unsigned char* pad = (const unsigned char*)d_in[2];
    const float* Wq  = (const float*)d_in[3];
    const float* bq  = (const float*)d_in[4];
    const float* Wk  = (const float*)d_in[5];
    const float* bk  = (const float*)d_in[6];
    const float* Wv  = (const float*)d_in[7];
    const float* bv  = (const float*)d_in[8];
    const float* Wo  = (const float*)d_in[9];
    const float* bo  = (const float*)d_in[10];
    const float* dw1 = (const float*)d_in[11];
    const float* db1 = (const float*)d_in[12];
    const float* dw2 = (const float*)d_in[13];
    const float* db2 = (const float*)d_in[14];
    const float* g1  = (const float*)d_in[15];
    const float* b1  = (const float*)d_in[16];
    const float* g2  = (const float*)d_in[17];
    const float* b2  = (const float*)d_in[18];
    const float* Wf1 = (const float*)d_in[19];
    const float* bf1 = (const float*)d_in[20];
    const float* Wf2 = (const float*)d_in[21];
    const float* bf2 = (const float*)d_in[22];

    const int M = M_ROWS;                      // 2048

    char* ws = (char*)d_ws;
    size_t off = 0;
    auto alloc = [&](size_t bytes) -> char* {
        char* p = ws + off;
        off = (off + bytes + 255) & ~(size_t)255;
        return p;
    };

    _Float16* wqT  = (_Float16*)alloc((size_t)D_MODEL * D_MODEL * 2);
    _Float16* wkT  = (_Float16*)alloc((size_t)D_MODEL * D_MODEL * 2);
    _Float16* wvT  = (_Float16*)alloc((size_t)D_MODEL * D_MODEL * 2);
    _Float16* woT  = (_Float16*)alloc((size_t)D_MODEL * D_MODEL * 2);
    _Float16* wf1T = (_Float16*)alloc((size_t)D_MODEL * D_FF * 2);
    _Float16* wf2T = (_Float16*)alloc((size_t)D_FF * D_MODEL * 2);
    _Float16* Q16 = (_Float16*)alloc((size_t)M * D_MODEL * 2);
    _Float16* K16 = (_Float16*)alloc((size_t)M * D_MODEL * 2);
    _Float16* Vt  = (_Float16*)alloc((size_t)M * D_MODEL * 2);   // [d][b*L+k]
    float* CTX = (float*)alloc((size_t)M * D_MODEL * 4);
    float* X2  = (float*)alloc((size_t)M * D_MODEL * 4);
    float* X1  = (float*)alloc((size_t)M * D_MODEL * 4);
    float* FF  = (float*)alloc((size_t)M * D_FF * 4);

    // weight transpose + convert (in: [K][N] f32 -> out: [N][K] f16)
    transp_f16<<<dim3(D_MODEL / 32, D_MODEL / 32), 256, 0, stream>>>(Wq,  wqT,  D_MODEL, D_MODEL);
    transp_f16<<<dim3(D_MODEL / 32, D_MODEL / 32), 256, 0, stream>>>(Wk,  wkT,  D_MODEL, D_MODEL);
    transp_f16<<<dim3(D_MODEL / 32, D_MODEL / 32), 256, 0, stream>>>(Wv,  wvT,  D_MODEL, D_MODEL);
    transp_f16<<<dim3(D_MODEL / 32, D_MODEL / 32), 256, 0, stream>>>(Wo,  woT,  D_MODEL, D_MODEL);
    transp_f16<<<dim3(D_FF / 32,    D_MODEL / 32), 256, 0, stream>>>(Wf1, wf1T, D_MODEL, D_FF);
    transp_f16<<<dim3(D_MODEL / 32, D_FF / 32),    256, 0, stream>>>(Wf2, wf2T, D_FF, D_MODEL);

    // QKV projections: Q,K f16 row-major; V f16 transposed
    gemm_wmma<<<dim3(D_MODEL / 64, M / 64), 128, 0, stream>>>(
        x, wqT, bq, nullptr, Q16, M, D_MODEL, D_MODEL, 0, 1);
    gemm_wmma<<<dim3(D_MODEL / 64, M / 64), 128, 0, stream>>>(
        x, wkT, bk, nullptr, K16, M, D_MODEL, D_MODEL, 0, 1);
    gemm_wmma<<<dim3(D_MODEL / 64, M / 64), 128, 0, stream>>>(
        x, wvT, bv, nullptr, Vt, M, D_MODEL, D_MODEL, 0, 2);

    // fused attention (flash-style, diff-MLP inline)
    attn_wmma<<<B_SZ * N_HEADS * (L_SEQ / 16), 32, 0, stream>>>(
        Q16, K16, Vt, mz, pad, dw1, db1, dw2, db2, CTX);

    // output projection + residual LN1
    gemm_wmma<<<dim3(D_MODEL / 64, M / 64), 128, 0, stream>>>(
        CTX, woT, bo, X2, nullptr, M, D_MODEL, D_MODEL, 0, 0);
    add_ln<<<M, 256, 0, stream>>>(x, X2, g1, b1, X1);

    // feedforward + residual LN2
    gemm_wmma<<<dim3(D_FF / 64, M / 64), 128, 0, stream>>>(
        X1, wf1T, bf1, FF, nullptr, M, D_FF, D_MODEL, 1, 0);
    gemm_wmma<<<dim3(D_MODEL / 64, M / 64), 128, 0, stream>>>(
        FF, wf2T, bf2, X2, nullptr, M, D_MODEL, D_FF, 0, 0);
    add_ln<<<M, 256, 0, stream>>>(X1, X2, g2, b2, (float*)d_out);
}
